// KMIL_3539053052016
// MI455X (gfx1250) — matrix operand order repaired and checked by reading, not compiled, not run on hardware
//
#include <hip/hip_runtime.h>
#include <math.h>

typedef __attribute__((ext_vector_type(16))) _Float16 v16h;
typedef __attribute__((ext_vector_type(8)))  float    v8f;
typedef __attribute__((ext_vector_type(4)))  float    f32x4;

#define B_      8
#define N_      8192
#define D_      512
#define H_      64
#define TOPK_   2457
#define JSPLIT_ 32
#define JCHUNK_ 77   /* ceil(2457/32) */

// Branch-free fast sigmoid/GELU: single v_exp_f32 + v_rcp_f32, no libm
// slow paths, no EXEC-mask divergence around WMMA regions.
__device__ __forceinline__ float fast_sigmoid(float x) {
  return __frcp_rn(1.0f + __expf(-x));
}
// tanh-approx GELU: 0.5x(1+tanh(u)) == x * sigmoid(2u), u = c(x+0.044715x^3)
__device__ __forceinline__ float gelu_tanh(float x) {
  float u = 0.7978845608028654f * (x + 0.044715f * x * x * x);
  return x * fast_sigmoid(2.0f * u);
}

__device__ __forceinline__ _Float16 u16_to_h(unsigned short u) {
  union { unsigned short s; _Float16 h; } c; c.s = u; return c.h;
}
__device__ __forceinline__ unsigned pack_f16x2(float a, float b) {
  union { _Float16 h; unsigned short s; } ca, cb;
  ca.h = (_Float16)a; cb.h = (_Float16)b;
  return (unsigned)ca.s | ((unsigned)cb.s << 16);
}

// ---------------------------------------------------------------------------
// Kernel 1: attention-score MLP for all B*N rows via WMMA f16.
// Wave = 16 rows; 16 K-steps of 16x16x32; 4 N-tiles (hidden=64).
// aW1 staged once per block into LDS as packed f16 K-pairs (B-fragment layout).
// ---------------------------------------------------------------------------
__global__ __launch_bounds__(256) void attn_score_kernel(
    const float* __restrict__ x, const float* __restrict__ aW1,
    const float* __restrict__ ab1, const float* __restrict__ aW2,
    const float* __restrict__ ab2, float* __restrict__ wout) {
  __shared__ unsigned sW1[256 * 64];  // dword [p*64+col] = {f16(aW1[2p,col]), f16(aW1[2p+1,col])}
  const int tid = threadIdx.x;
  for (int i = tid; i < 256 * 64; i += 256) {
    int p = i >> 6, col = i & 63;
    sW1[i] = pack_f16x2(aW1[(2 * p) * H_ + col], aW1[(2 * p + 1) * H_ + col]);
  }
  __syncthreads();

  const int wave = tid >> 5;
  const int lane = tid & 31;
  const int m  = lane & 15;   // row within tile / N-column within B tile
  const int hi = lane >> 4;   // half-wave selector for K
  const long rowBase = (long)blockIdx.x * 128 + (long)wave * 16;
  const float* xrow = x + (rowBase + m) * (long)D_;

  v8f acc[4] = {};
  for (int kt = 0; kt < 16; ++kt) {
    // A fragment: lane m/hi holds K = kt*32 + hi*8 + {0..7} and +16..23
    const int kb = kt * 32 + hi * 8;
    f32x4 f0 = *(const f32x4*)(xrow + kb);
    f32x4 f1 = *(const f32x4*)(xrow + kb + 4);
    f32x4 f2 = *(const f32x4*)(xrow + kb + 16);
    f32x4 f3 = *(const f32x4*)(xrow + kb + 20);
    v16h a;
#pragma unroll
    for (int e = 0; e < 4; ++e) {
      a[e]      = (_Float16)f0[e];
      a[4 + e]  = (_Float16)f1[e];
      a[8 + e]  = (_Float16)f2[e];
      a[12 + e] = (_Float16)f3[e];
    }
    const int pB = kt * 16 + hi * 4;  // K-pair base for B fragment
#pragma unroll
    for (int t = 0; t < 4; ++t) {
      v16h bb;
#pragma unroll
      for (int v = 0; v < 8; ++v) {
        unsigned d = sW1[(pB + v + (v >= 4 ? 4 : 0)) * 64 + t * 16 + m];
        bb[2 * v]     = u16_to_h((unsigned short)(d & 0xffffu));
        bb[2 * v + 1] = u16_to_h((unsigned short)(d >> 16));
      }
      acc[t] = __builtin_amdgcn_wmma_f32_16x16x32_f16(
          false, a, false, bb, (short)0, acc[t], false, false);
    }
  }

  // epilogue: h = GELU(acc + ab1[col]); score = h . aW2 + ab2; sigmoid.
  float p[8] = {0, 0, 0, 0, 0, 0, 0, 0};
#pragma unroll
  for (int t = 0; t < 4; ++t) {
    float b1 = ab1[t * 16 + m];
    float w2 = aW2[t * 16 + m];
#pragma unroll
    for (int r = 0; r < 8; ++r)
      p[r] += gelu_tanh(acc[t][r] + b1) * w2;
  }
#pragma unroll
  for (int r = 0; r < 8; ++r) {  // reduce across the 16-lane half (rows r+8*hi)
    p[r] += __shfl_xor(p[r], 1);
    p[r] += __shfl_xor(p[r], 2);
    p[r] += __shfl_xor(p[r], 4);
    p[r] += __shfl_xor(p[r], 8);
  }
  if (m == 0) {
    float bias2 = ab2[0];
    long rb = rowBase + 8 * hi;
#pragma unroll
    for (int r = 0; r < 8; ++r)
      wout[rb + r] = fast_sigmoid(p[r] + bias2);
  }
}

// ---------------------------------------------------------------------------
// Kernel 2: exact top-k threshold per bag (bisection on positive-float bits)
// + deterministic prefix-sum compaction (stable, no atomics).
// ---------------------------------------------------------------------------
__global__ __launch_bounds__(256) void topk_select_kernel(
    const float* __restrict__ wout, int* __restrict__ idxbuf,
    float* __restrict__ wselbuf) {
  __shared__ unsigned su[N_];
  __shared__ unsigned red[256];
  __shared__ unsigned scg[256], sce[256];
  __shared__ unsigned sTot;
  const int b = blockIdx.x;
  const int tid = threadIdx.x;
  const float* wb = wout + (long)b * N_;
  for (int i = tid; i < N_; i += 256)
    su[i] = __builtin_bit_cast(unsigned, wb[i]);  // sigmoid > 0: bits order = value order
  __syncthreads();

  const int base = tid * 32;
  unsigned lo = 0u, hiT = 0x40000000u;  // sigmoid <= 1.0f = 0x3F800000
  while (hiT - lo > 1u) {
    unsigned mid = lo + ((hiT - lo) >> 1);
    unsigned c = 0;
    for (int i = 0; i < 32; ++i) c += (su[base + i] >= mid) ? 1u : 0u;
    red[tid] = c;
    __syncthreads();
    for (int s2 = 128; s2 > 0; s2 >>= 1) {
      if (tid < s2) red[tid] += red[tid + s2];
      __syncthreads();
    }
    unsigned tot = red[0];
    __syncthreads();
    if (tot >= TOPK_) lo = mid; else hiT = mid;
  }
  const unsigned thr = lo;  // count(>= thr) >= K, count(> thr) < K

  unsigned cg = 0, ce = 0;
  for (int i = 0; i < 32; ++i) {
    unsigned u = su[base + i];
    if (u > thr) cg++;
    else if (u == thr) ce++;
  }
  scg[tid] = cg; sce[tid] = ce;
  __syncthreads();
  if (tid == 0) {  // serial exclusive scan (256 entries, trivial)
    unsigned ag = 0, ae = 0;
    for (int t = 0; t < 256; ++t) {
      unsigned tg = scg[t]; scg[t] = ag; ag += tg;
      unsigned te = sce[t]; sce[t] = ae; ae += te;
    }
    sTot = ag;
  }
  __syncthreads();
  const unsigned totGT = sTot;
  unsigned og = scg[tid], oe = sce[tid];
  int* ib = idxbuf + (long)b * TOPK_;
  float* fb = wselbuf + (long)b * TOPK_;
  for (int i = 0; i < 32; ++i) {
    int n = base + i;
    unsigned u = su[n];
    if (u > thr) {
      ib[og] = n; fb[og] = __builtin_bit_cast(float, u); og++;
    } else if (u == thr) {
      unsigned pos = totGT + oe;
      if (pos < TOPK_) { ib[pos] = n; fb[pos] = __builtin_bit_cast(float, u); }
      oe++;
    }
  }
}

// ---------------------------------------------------------------------------
// Kernel 3: weighted gather-accumulate, split over 32 j-chunks per bag
// into a fixed-order partial buffer (deterministic, no float atomics).
// Second pass over x rides the 192 MB L2 (x = 128 MiB, still resident).
// ---------------------------------------------------------------------------
__global__ __launch_bounds__(256) void gather_accum_kernel(
    const float* __restrict__ x, const int* __restrict__ idxbuf,
    const float* __restrict__ wselbuf, float* __restrict__ partial) {
  const int b = blockIdx.x & (B_ - 1);
  const int s = blockIdx.x >> 3;
  const int tid = threadIdx.x;
  const int j0 = s * JCHUNK_;
  int j1 = j0 + JCHUNK_; if (j1 > TOPK_) j1 = TOPK_;
  float a0 = 0.f, a1 = 0.f;
  const int* ib = idxbuf + (long)b * TOPK_;
  const float* fb = wselbuf + (long)b * TOPK_;
  for (int j = j0; j < j1; ++j) {
    int i = ib[j];
    float w = fb[j];
    const float* row = x + ((long)b * N_ + i) * D_;
    a0 += w * row[tid];
    a1 += w * row[tid + 256];
  }
  float* pr = partial + (long)(s * B_ + b) * D_;
  pr[tid] = a0;
  pr[tid + 256] = a1;
}

__global__ __launch_bounds__(256) void reduce_emb_kernel(
    const float* __restrict__ partial, float* __restrict__ emb) {
  int g = blockIdx.x * 256 + threadIdx.x;  // 0..4095
  int b = g >> 9, d = g & 511;
  float s = 0.f;
  for (int sp = 0; sp < JSPLIT_; ++sp)
    s += partial[(long)(sp * B_ + b) * D_ + d];
  emb[g] = s * (1.0f / (float)TOPK_);
}

// ---------------------------------------------------------------------------
// Kernel 4: projection MLP (8x512 @ 512x512 -> GELU -> @ 512x512) via WMMA,
// 8 rows padded to one 16-row tile, single block, hidden staged in LDS f16.
// ---------------------------------------------------------------------------
__global__ __launch_bounds__(256) void proj_mlp_kernel(
    const float* __restrict__ emb, const float* __restrict__ pW1,
    const float* __restrict__ pb1, const float* __restrict__ pW2,
    const float* __restrict__ pb2, float* __restrict__ out) {
  __shared__ _Float16 sA[16 * 512];
  __shared__ _Float16 sH[16 * 512];
  const int tid = threadIdx.x;
  for (int i = tid; i < 16 * 512; i += 256) {
    int mrow = i >> 9;
    sA[i] = (_Float16)((mrow < 8) ? emb[i] : 0.0f);
  }
  __syncthreads();

  const int wv  = tid >> 5;
  const int lane = tid & 31;
  const int nn  = lane & 15;
  const int hhi = lane >> 4;

  v8f acc[4] = {};
  for (int kt = 0; kt < 16; ++kt) {
    v16h a;
    const int abase = nn * 512 + kt * 32 + hhi * 8;
#pragma unroll
    for (int e = 0; e < 8; ++e) { a[e] = sA[abase + e]; a[8 + e] = sA[abase + 16 + e]; }
#pragma unroll
    for (int t = 0; t < 4; ++t) {
      const int col = wv * 64 + t * 16 + nn;
      v16h bb;
#pragma unroll
      for (int e = 0; e < 16; ++e) {
        int K = kt * 32 + hhi * 8 + (e < 8 ? e : e + 8);
        bb[e] = (_Float16)pW1[K * 512 + col];
      }
      acc[t] = __builtin_amdgcn_wmma_f32_16x16x32_f16(
          false, a, false, bb, (short)0, acc[t], false, false);
    }
  }
#pragma unroll
  for (int t = 0; t < 4; ++t) {
    const int col = wv * 64 + t * 16 + nn;
    float b1 = pb1[col];
#pragma unroll
    for (int r = 0; r < 8; ++r)
      sH[(r + 8 * hhi) * 512 + col] = (_Float16)gelu_tanh(acc[t][r] + b1);
  }
  __syncthreads();

  v8f acc2[4] = {};
  for (int kt = 0; kt < 16; ++kt) {
    v16h a;
    const int abase = nn * 512 + kt * 32 + hhi * 8;
#pragma unroll
    for (int e = 0; e < 8; ++e) { a[e] = sH[abase + e]; a[8 + e] = sH[abase + 16 + e]; }
#pragma unroll
    for (int t = 0; t < 4; ++t) {
      const int col = wv * 64 + t * 16 + nn;
      v16h bb;
#pragma unroll
      for (int e = 0; e < 16; ++e) {
        int K = kt * 32 + hhi * 8 + (e < 8 ? e : e + 8);
        bb[e] = (_Float16)pW2[K * 512 + col];
      }
      acc2[t] = __builtin_amdgcn_wmma_f32_16x16x32_f16(
          false, a, false, bb, (short)0, acc2[t], false, false);
    }
  }
  if (hhi == 0) {  // rows 0..7 are the real batch rows
#pragma unroll
    for (int t = 0; t < 4; ++t) {
      const int col = wv * 64 + t * 16 + nn;
      float b2 = pb2[col];
#pragma unroll
      for (int r = 0; r < 8; ++r)
        out[r * 512 + col] = acc2[t][r] + b2;
    }
  }
}

// ---------------------------------------------------------------------------
extern "C" void kernel_launch(void* const* d_in, const int* in_sizes, int n_in,
                              void* d_out, int out_size, void* d_ws, size_t ws_size,
                              hipStream_t stream) {
  (void)in_sizes; (void)n_in; (void)out_size; (void)ws_size;
  const float* x   = (const float*)d_in[0];
  const float* aW1 = (const float*)d_in[1];
  const float* ab1 = (const float*)d_in[2];
  const float* aW2 = (const float*)d_in[3];
  const float* ab2 = (const float*)d_in[4];
  const float* pW1 = (const float*)d_in[5];
  const float* pb1 = (const float*)d_in[6];
  const float* pW2 = (const float*)d_in[7];
  const float* pb2 = (const float*)d_in[8];

  float* out_proj = (float*)d_out;          // [8,512]
  float* out_w    = out_proj + B_ * D_;     // [8,8192]

  char* ws = (char*)d_ws;
  int*   ws_idx  = (int*)ws;                                        // 8*2457 ints
  float* ws_sel  = (float*)(ws + (size_t)B_ * TOPK_ * 4);           // 8*2457 floats
  float* ws_part = (float*)(ws + (size_t)2 * B_ * TOPK_ * 4);       // 32*8*512 floats
  float* ws_emb  = ws_part + (size_t)JSPLIT_ * B_ * D_;             // 8*512 floats

  attn_score_kernel<<<(B_ * N_) / 128, 256, 0, stream>>>(x, aW1, ab1, aW2, ab2, out_w);
  topk_select_kernel<<<B_, 256, 0, stream>>>(out_w, ws_idx, ws_sel);
  gather_accum_kernel<<<B_ * JSPLIT_, 256, 0, stream>>>(x, ws_idx, ws_sel, ws_part);
  reduce_emb_kernel<<<(B_ * D_) / 256, 256, 0, stream>>>(ws_part, ws_emb);
  proj_mlp_kernel<<<1, 256, 0, stream>>>(ws_emb, pW1, pb1, pW2, pb2, out_proj);
}
